// ParallelExperts_5592047419559
// MI455X (gfx1250) — compile-verified, matrix operand order checked by reading, not run on hardware
//
#include <hip/hip_runtime.h>
#include <hip/hip_bf16.h>
#include <stdint.h>

// ---------------- problem constants (static per reference) ----------------
#define NTOK   4096
#define KFAN   2
#define NEXP   8
#define DIN    1024
#define DOUT   1024
#define MTOT   (NTOK * KFAN)        // 8192 grouped rows

// ---------------- tile config ----------------
#define BM     128
#define BN     128
#define BK     64                   // two 16x16x32 K-steps per staged tile
#define LDA    72                   // padded LDS row stride in bf16 (144 B)
#define KTILES (DIN / BK)           // 16
#define ABUF_B (BM * LDA * 2)       // bytes per A LDS buffer (18432)
#define BBUF_B (BN * LDA * 2)       // bytes per B LDS buffer (18432)

typedef __attribute__((ext_vector_type(8)))  float        v8f;
typedef __attribute__((ext_vector_type(16))) __bf16       v16bf;
typedef __attribute__((ext_vector_type(8)))  __bf16       bf16x8;
typedef __attribute__((ext_vector_type(4)))  unsigned int u32x4;
typedef __attribute__((ext_vector_type(8)))  int          i32x8;

// ---------------- prep kernels ----------------
__global__ void zero_f32(float* p, int n) {
    int i = blockIdx.x * blockDim.x + threadIdx.x;
    if (i < n) p[i] = 0.0f;
}

__global__ void cvt_f32_bf16(const float* __restrict__ src,
                             __hip_bfloat16* __restrict__ dst, int n) {
    int i = blockIdx.x * blockDim.x + threadIdx.x;
    if (i < n) dst[i] = __float2bfloat16(src[i]);
}

// ---------------- async copy: global -> LDS (per-lane gather, ASYNCcnt) ----------------
__device__ __forceinline__ void async_copy_b128(uint32_t lds_off, uint64_t gaddr) {
    asm volatile("global_load_async_to_lds_b128 %0, %1, off"
                 :: "v"(lds_off), "v"(gaddr) : "memory");
}
__device__ __forceinline__ void wait_async0() {
    asm volatile("s_wait_asynccnt 0" ::: "memory");
}

// ---------------- TDM: dense B tile global -> LDS (TENSORcnt) ----------------
// 2D tile: BM rows x BK bf16 elements, row stride DIN elements in memory,
// hardware-padded in LDS: 128B data + 16B pad per row  ->  144B = LDA*2 stride.
__device__ __forceinline__ void tdm_load_b_tile(uint32_t lds_addr, uint64_t gaddr) {
    u32x4 g0;
    g0[0] = 1u;                                            // count=1 (valid, user)
    g0[1] = lds_addr;                                      // [63:32]  lds_addr
    g0[2] = (uint32_t)gaddr;                               // [95:64]  global_addr lo
    g0[3] = (uint32_t)((gaddr >> 32) & 0x01FFFFFFu)        // [120:96] global_addr hi
          | 0x80000000u;                                   // [127:126] type=2 (image)
    i32x8 g1;
    g1[0] = (1 << 16)                                      // data_size = 2 bytes
          | (1 << 20)                                      // pad_enable
          | (4 << 22)                                      // pad_interval: 32 DWORDs (128B)
          | (3 << 25);                                     // pad_amount:   4 DWORDs (16B)
    g1[1] = (DIN & 0xFFFF) << 16;                          // tensor_dim0[15:0]  -> [63:48]
    g1[2] = (DOUT & 0xFFFF) << 16;                         // tensor_dim1[15:0]  -> [95:80]
    g1[3] = BK << 16;                                      // tile_dim0 = 64     -> [127:112]
    g1[4] = BM;                                            // tile_dim1 = 128    -> [143:128]
    g1[5] = DIN;                                           // tensor_dim0_stride lo32
    g1[6] = 0;
    g1[7] = 0;
    asm volatile("tensor_load_to_lds %0, %1" :: "s"(g0), "s"(g1) : "memory");
}

// ---------------- grouped GEMM + gated scatter-combine ----------------
__global__ __launch_bounds__(256) void moe_gemm_bf16(
    const __bf16* __restrict__ xb,      // [NTOK, DIN] bf16
    const __bf16* __restrict__ wb,      // [NEXP, DOUT, DIN] bf16
    const float*  __restrict__ gates,   // [NTOK, KFAN]
    const int*    __restrict__ seid,    // sorted_expert_idxs [MTOT]
    const int*    __restrict__ ssid,    // sorted_scattered_idxs [MTOT]
    float*        __restrict__ out)     // [NTOK, DOUT], pre-zeroed
{
    __shared__ __bf16 As[2][BM * LDA];
    __shared__ __bf16 Bs[2][BN * LDA];
    __shared__ int s_eid[BM];
    __shared__ int s_tok[BM];
    __shared__ int s_sid[BM];

    const int tid = threadIdx.x;
    const int m0  = blockIdx.y * BM;
    const int n0  = blockIdx.x * BN;

    if (tid < BM) {
        int m = m0 + tid;
        int sid = ssid[m];
        s_eid[tid] = seid[m];
        s_sid[tid] = sid;
        s_tok[tid] = sid >> 1;          // KFAN == 2
    }
    __syncthreads();

    const int e_first = s_eid[0];
    const int e_last  = s_eid[BM - 1];

    const int wave = tid >> 5;          // 0..7 (wave32)
    const int lane = tid & 31;
    const int wm   = wave >> 2;         // 0..1 : 64 rows each
    const int wn   = wave & 3;          // 0..3 : 32 cols each
    const int lrow = lane & 15;
    const int lhi  = lane >> 4;         // K-half select per ISA 16-bit layout

    // A staging: 256 threads x 4 rows x one 16B chunk
    const int srow = tid >> 3;          // 0..31
    const int scol = (tid & 7) * 8;     // bf16 element offset (16 B chunks)

    // loop-invariant A gather addresses (token row bases) and LDS slots
    uint64_t gaA[4];
    uint32_t laA[4];
#pragma unroll
    for (int p = 0; p < 4; ++p) {
        int r = srow + 32 * p;
        gaA[p] = (uint64_t)(uintptr_t)(xb + (size_t)s_tok[r] * DIN + scol);
        laA[p] = (uint32_t)(size_t)&As[0][r * LDA + scol];
    }
    const uint32_t ldsB0 = (uint32_t)(size_t)&Bs[0][0];

    for (int e = e_first; e <= e_last; ++e) {
        const size_t wbase = (size_t)e * DOUT * DIN;
        const uint64_t gB  = (uint64_t)(uintptr_t)(wb + wbase + (size_t)n0 * DIN);

        v8f acc[4][2];
#pragma unroll
        for (int i = 0; i < 4; ++i)
#pragma unroll
            for (int j = 0; j < 2; ++j)
                acc[i][j] = (v8f){};

        // ---- prologue: stage k-tile 0 into buffer 0 ----
        if (wave == 0)
            tdm_load_b_tile(ldsB0, gB);
#pragma unroll
        for (int p = 0; p < 4; ++p)
            async_copy_b128(laA[p], gaA[p]);
        if (wave == 0)
            __builtin_amdgcn_s_wait_tensorcnt(0);
        wait_async0();
        __syncthreads();

        for (int kt = 0; kt < KTILES; ++kt) {
            const int cur = kt & 1;
            const int nxt = cur ^ 1;

            // ---- issue async + TDM stage of next k-tile (no wait yet) ----
            if (kt + 1 < KTILES) {
                const int k0 = (kt + 1) * BK;
                if (wave == 0)
                    tdm_load_b_tile(ldsB0 + (uint32_t)nxt * BBUF_B,
                                    gB + (uint64_t)(k0 * 2));
#pragma unroll
                for (int p = 0; p < 4; ++p)
                    async_copy_b128(laA[p] + (uint32_t)nxt * ABUF_B,
                                    gaA[p] + (uint64_t)(k0 * 2));
            }

            // ---- compute on current buffer: two 16x16x32 K-steps ----
#pragma unroll
            for (int kk = 0; kk < BK; kk += 32) {
                v16bf afrag[4], bfrag[2];
#pragma unroll
                for (int i = 0; i < 4; ++i) {
                    const __bf16* ap =
                        &As[cur][(wm * 64 + i * 16 + lrow) * LDA + kk + lhi * 8];
                    bf16x8 lo = *(const bf16x8*)ap;
                    bf16x8 hi = *(const bf16x8*)(ap + 16);
                    afrag[i] = __builtin_shufflevector(lo, hi,
                        0,1,2,3,4,5,6,7,8,9,10,11,12,13,14,15);
                }
#pragma unroll
                for (int j = 0; j < 2; ++j) {
                    const __bf16* bp =
                        &Bs[cur][(wn * 32 + j * 16 + lrow) * LDA + kk + lhi * 8];
                    bf16x8 lo = *(const bf16x8*)bp;
                    bf16x8 hi = *(const bf16x8*)(bp + 16);
                    bfrag[j] = __builtin_shufflevector(lo, hi,
                        0,1,2,3,4,5,6,7,8,9,10,11,12,13,14,15);
                }
#pragma unroll
                for (int i = 0; i < 4; ++i)
#pragma unroll
                    for (int j = 0; j < 2; ++j)
                        acc[i][j] = __builtin_amdgcn_wmma_f32_16x16x32_bf16(
                            false, afrag[i], false, bfrag[j],
                            (short)0, acc[i][j], false, false);
            }

            // next buffers landed; everyone done reading current buffers
            if (wave == 0)
                __builtin_amdgcn_s_wait_tensorcnt(0);
            wait_async0();
            __syncthreads();
        }

        // ---- epilogue: gated scatter back to token rows ----
#pragma unroll
        for (int i = 0; i < 4; ++i) {
#pragma unroll
            for (int r = 0; r < 8; ++r) {
                const int rloc = wm * 64 + i * 16 + lhi * 8 + r;
                if (s_eid[rloc] == e) {
                    const int   sid  = s_sid[rloc];
                    const int   tok  = sid >> 1;
                    const int   slot = sid & 1;
                    const float g    = gates[tok * KFAN + slot];
                    float* orow = out + (size_t)tok * DOUT;
#pragma unroll
                    for (int j = 0; j < 2; ++j) {
                        const int ncol = n0 + wn * 32 + j * 16 + lrow;
                        atomicAdd(orow + ncol, g * acc[i][j][r]);
                    }
                }
            }
        }
        __syncthreads();   // LDS buffers reused by next expert iteration
    }
}

// ---------------- host launcher ----------------
extern "C" void kernel_launch(void* const* d_in, const int* in_sizes, int n_in,
                              void* d_out, int out_size, void* d_ws, size_t ws_size,
                              hipStream_t stream) {
    const float* x     = (const float*)d_in[0];   // [NTOK, DIN]
    const float* w     = (const float*)d_in[1];   // [NEXP, DOUT, DIN]
    const float* gates = (const float*)d_in[2];   // [NTOK, KFAN]
    const int*   seid  = (const int*)d_in[3];     // [MTOT]
    const int*   ssid  = (const int*)d_in[4];     // [MTOT]
    float*       out   = (float*)d_out;           // [NTOK, DOUT]

    __hip_bfloat16* xb_h = (__hip_bfloat16*)d_ws;                  //  8 MB
    __hip_bfloat16* wb_h = xb_h + (size_t)NTOK * DIN;              // 16 MB

    const int nx = NTOK * DIN;          // 4 Mi elements
    const int nw = NEXP * DOUT * DIN;   // 8 Mi elements
    const int no = NTOK * DOUT;         // 4 Mi elements

    zero_f32    <<<(no + 255) / 256, 256, 0, stream>>>(out, no);
    cvt_f32_bf16<<<(nx + 255) / 256, 256, 0, stream>>>(x, xb_h, nx);
    cvt_f32_bf16<<<(nw + 255) / 256, 256, 0, stream>>>(w, wb_h, nw);

    dim3 grid(DOUT / BN, MTOT / BM);    // (8, 64)
    moe_gemm_bf16<<<grid, 256, 0, stream>>>(
        (const __bf16*)xb_h, (const __bf16*)wb_h, gates, seid, ssid, out);
}
